// CPM3Attention_51075751084393
// MI455X (gfx1250) — matrix operand order repaired and checked by reading, not compiled, not run on hardware
//
#include <hip/hip_runtime.h>
#include <hip/hip_bf16.h>

// CPM3 attention for MI455X (gfx1250, wave32, WMMA f32_16x16x32_f16).
// Pipeline: [f32->f16 convert] -> [QKV proj GEMM, 32x64/wave] ->
//           [flash attention w/ double-buffered async LDS staging] -> [out proj GEMM]

typedef __attribute__((ext_vector_type(16))) _Float16 v16h;
typedef __attribute__((ext_vector_type(8)))  _Float16 h8;
typedef __attribute__((ext_vector_type(2)))  _Float16 h2;
typedef __attribute__((ext_vector_type(8)))  float    v8f;
typedef __attribute__((ext_vector_type(4)))  float    f4;
typedef __attribute__((ext_vector_type(4)))  int      i4;

#define B_  2
#define LQ_ 2048
#define LK_ 2048
#define D_  1024
#define H_  16
#define DH_ 64

__device__ __forceinline__ v8f wmma_f16(v16h a, v16h b, v8f c) {
  // (neg_a, A, neg_b, B, c_mod, C, reuse_a, reuse_b)
  return __builtin_amdgcn_wmma_f32_16x16x32_f16(false, a, false, b, (short)0, c, false, false);
}

// A-operand: elements 0..7 = low K-run, 8..15 = high K-run (ISA 16-bit A layout).
__device__ __forceinline__ v16h pack_h8(h8 lo, h8 hi) {
  v16h a;
#pragma unroll
  for (int i = 0; i < 8; ++i) { a[i] = lo[i]; a[i + 8] = hi[i]; }
  return a;
}

__device__ __forceinline__ v16h cvt_f32x16(f4 x0, f4 x1, f4 x2, f4 x3) {
  v16h v;
#pragma unroll
  for (int i = 0; i < 4; ++i) {
    v[i]      = (_Float16)x0[i];
    v[i + 4]  = (_Float16)x1[i];
    v[i + 8]  = (_Float16)x2[i];
    v[i + 12] = (_Float16)x3[i];
  }
  return v;
}

// ---------------------------------------------------------------------------
// Kernel 0: bulk f32 -> f16 convert (16 elements / thread).
// ---------------------------------------------------------------------------
__global__ void __launch_bounds__(256)
cpm3_cvt_f16(const float* __restrict__ src, _Float16* __restrict__ dst, int n16) {
  const int i = blockIdx.x * blockDim.x + threadIdx.x;
  if (i < n16) {
    const f4* s = (const f4*)src + (size_t)i * 4;
    *(v16h*)(dst + (size_t)i * 16) = cvt_f32x16(s[0], s[1], s[2], s[3]);
  }
}

// ---------------------------------------------------------------------------
// Kernel 1: QKV projections from pre-converted f16 X and W.
// mode = blockIdx.z: 0 -> Qh, 1 -> Kh, 2 -> Vt (V transposed: [B,H,64,LK]).
// One wave owns a 32x64 C block: 8 WMMAs per 12 b128 loads per K=32 step.
// ---------------------------------------------------------------------------
__global__ void __launch_bounds__(256)
cpm3_qkv_proj(const _Float16* __restrict__ Xq, const _Float16* __restrict__ Xkv,
              const _Float16* __restrict__ Wq16, const _Float16* __restrict__ Wk16,
              const _Float16* __restrict__ Wv16,
              _Float16* __restrict__ Qh, _Float16* __restrict__ Kh,
              _Float16* __restrict__ Vt) {
  const int mode = blockIdx.z;
  const _Float16* __restrict__ X = (mode == 0) ? Xq : Xkv;
  const _Float16* __restrict__ W = (mode == 0) ? Wq16 : ((mode == 1) ? Wk16 : Wv16);

  const int lane = threadIdx.x & 31;
  const int wave = threadIdx.x >> 5;
  const int l16  = lane & 15;
  const int hf   = lane >> 4;

  const int tile = blockIdx.x * 8 + wave;   // 2048 tiles: 128 Mb x 16 Nb
  const int m0 = (tile >> 4) * 32;
  const int n0 = (tile & 15) * 64;

  const _Float16* __restrict__ Arow0 = X + (size_t)(m0 + l16) * D_;
  const _Float16* __restrict__ Arow1 = X + (size_t)(m0 + 16 + l16) * D_;
  const _Float16* __restrict__ Brow  = W + (size_t)(n0 + l16) * D_;

  v8f acc[2][4] = {};
#pragma unroll 2
  for (int kk = 0; kk < D_ / 32; ++kk) {
    const int ka = kk * 32 + hf * 16;
    const v16h A0 = pack_h8(*(const h8*)(Arow0 + ka), *(const h8*)(Arow0 + ka + 8));
    const v16h A1 = pack_h8(*(const h8*)(Arow1 + ka), *(const h8*)(Arow1 + ka + 8));
    const int kb = kk * 32 + hf * 16;
#pragma unroll
    for (int ni = 0; ni < 4; ++ni) {
      const v16h Bv = *(const v16h*)(Brow + (size_t)ni * 16 * D_ + kb);
      acc[0][ni] = wmma_f16(A0, Bv, acc[0][ni]);
      acc[1][ni] = wmma_f16(A1, Bv, acc[1][ni]);
    }
  }

#pragma unroll
  for (int mi = 0; mi < 2; ++mi) {
#pragma unroll
    for (int ni = 0; ni < 4; ++ni) {
      const int n = n0 + ni * 16 + l16;
      const int h = n >> 6;
      const int d = n & 63;
#pragma unroll
      for (int r = 0; r < 8; ++r) {
        const int m  = m0 + mi * 16 + hf * 8 + r;
        const int bb = m >> 11;
        const int l  = m & (LQ_ - 1);
        const _Float16 v = (_Float16)acc[mi][ni][r];
        if (mode == 0)      Qh[(((size_t)bb * H_ + h) * LQ_ + l) * DH_ + d] = v;
        else if (mode == 1) Kh[(((size_t)bb * H_ + h) * LK_ + l) * DH_ + d] = v;
        else                Vt[(((size_t)bb * H_ + h) * DH_ + d) * LK_ + l] = v;
      }
    }
  }
}

// ---------------------------------------------------------------------------
// Kernel 2: flash attention.  Block = (128 q rows, b, h); 8 waves x 16 q rows.
// K/V 32-k tiles staged global->LDS with CDNA5 async-LDS loads, ping-pong
// double buffered: tile t+1 is in flight (ASYNCcnt=2) while tile t computes.
// S^T = K.Q^T (softmax over k is lane-local + one shfl_xor(16)),
// ctx^T += V^T . P^T with online-softmax rescaling.
// ---------------------------------------------------------------------------
__global__ void __launch_bounds__(256)
cpm3_attn(const _Float16* __restrict__ Qh, const _Float16* __restrict__ Kh,
          const _Float16* __restrict__ Vt, const float* __restrict__ bias,
          const int* __restrict__ mask, _Float16* __restrict__ ctx) {
  __shared__ _Float16 Ktile[2][32 * 64];   // [buf][k][d]
  __shared__ _Float16 Vtile[2][64 * 32];   // [buf][d][k]
  const int NT = LK_ / 32;

  const int bh = blockIdx.y;            // b*H + h
  const int b  = bh >> 4;
  const int h  = bh & 15;
  const int lane = threadIdx.x & 31;
  const int wave = threadIdx.x >> 5;
  const int l16 = lane & 15;
  const int hf  = lane >> 4;
  const int q   = blockIdx.x * 128 + wave * 16 + l16;

  const _Float16* Qrow = Qh + ((size_t)bh * LQ_ + q) * DH_;
  const v16h qB0 = pack_h8(*(const h8*)(Qrow + hf * 8), *(const h8*)(Qrow + hf * 8 + 16));
  const v16h qB1 = pack_h8(*(const h8*)(Qrow + hf * 8 + 32), *(const h8*)(Qrow + hf * 8 + 48));

  const _Float16* Kbase = Kh + (size_t)bh * LK_ * DH_;
  const _Float16* Vbase = Vt + (size_t)bh * DH_ * LK_;
  const float*    biasQ = bias + ((size_t)h * LQ_ + q) * LK_;
  const int*      maskQ = mask + ((size_t)b * LQ_ + q) * LK_;

  // cooperative staging coordinates (256 threads, 16B each)
  const int kr = threadIdx.x >> 3, kc = (threadIdx.x & 7) * 8;   // K: 32 x 64
  const int vr = threadIdx.x >> 2, vc = (threadIdx.x & 3) * 8;   // V: 64 x 32
  // LDS byte offsets (low 32 bits of the flat LDS-aperture address).
  unsigned kdst[2], vdst[2];
  kdst[0] = (unsigned)(uintptr_t)&Ktile[0][kr * 64 + kc];
  kdst[1] = (unsigned)(uintptr_t)&Ktile[1][kr * 64 + kc];
  vdst[0] = (unsigned)(uintptr_t)&Vtile[0][vr * 32 + vc];
  vdst[1] = (unsigned)(uintptr_t)&Vtile[1][vr * 32 + vc];

  auto issue_tile = [&](int kb2, int bufi) {
    const _Float16* ksrc = Kbase + (size_t)(kb2 + kr) * DH_ + kc;
    const _Float16* vsrc = Vbase + (size_t)vr * LK_ + kb2 + vc;
    asm volatile("global_load_async_to_lds_b128 %0, %1, off"
                 :: "v"(kdst[bufi]), "v"(ksrc) : "memory");
    asm volatile("global_load_async_to_lds_b128 %0, %1, off"
                 :: "v"(vdst[bufi]), "v"(vsrc) : "memory");
  };

  float m_run = -3.0e30f;
  float l_run = 0.0f;
  v8f acc[4] = {};   // ctx^T: 64 d-rows x 16 q-cols

  issue_tile(0, 0);

  for (int kt = 0; kt < NT; ++kt) {
    const int kb = kt * 32;
    const int buf = kt & 1;

    if (kt + 1 < NT) {
      // Buffer (kt+1)&1 was released by the barrier ending iteration kt-1.
      issue_tile(kb + 32, buf ^ 1);
      if (kt + 2 < NT) {
        __builtin_prefetch(Kbase + (size_t)(kb + 64 + kr) * DH_ + kc, 0, 0);
        __builtin_prefetch(Vbase + (size_t)vr * LK_ + kb + 64 + vc, 0, 0);
      }
      // Async loads complete in order: <=2 outstanding means tile kt is in LDS.
      asm volatile("s_wait_asynccnt 0x2" ::: "memory");
    } else {
      asm volatile("s_wait_asynccnt 0x0" ::: "memory");
    }
    __syncthreads();

    const _Float16* Kt = Ktile[buf];
    const _Float16* Vw = Vtile[buf];

    // S^T (32k x 16q) as two 16x16 tiles; lane holds k = r + hf*8 (+16*t2), col q.
    float p0[8], p1[8];
    float mloc = -3.0e30f;
#pragma unroll
    for (int t2 = 0; t2 < 2; ++t2) {
      v8f s = {};
      const _Float16* krow = &Kt[(t2 * 16 + l16) * 64 + hf * 8];
      s = wmma_f16(pack_h8(*(const h8*)(krow),      *(const h8*)(krow + 16)), qB0, s);
      s = wmma_f16(pack_h8(*(const h8*)(krow + 32), *(const h8*)(krow + 48)), qB1, s);

      const int kk = kb + t2 * 16 + hf * 8;
      const f4 bq0 = *(const f4*)(biasQ + kk);
      const f4 bq1 = *(const f4*)(biasQ + kk + 4);
      const i4 mq0 = *(const i4*)(maskQ + kk);
      const i4 mq1 = *(const i4*)(maskQ + kk + 4);
      float* pt = t2 ? p1 : p0;
#pragma unroll
      for (int r = 0; r < 8; ++r) {
        float sv = s[r] * 0.125f + ((r < 4) ? bq0[r] : bq1[r - 4]);
        const int mv = (r < 4) ? mq0[r] : mq1[r - 4];
        sv = mv ? sv : -3.0e30f;
        pt[r] = sv;
        mloc = fmaxf(mloc, sv);
      }
    }

    mloc = fmaxf(mloc, __shfl_xor(mloc, 16, 32));
    const float m_new   = fmaxf(m_run, mloc);
    const float rescale = __expf(m_run - m_new);
    m_run = m_new;
    float lsum = 0.0f;
#pragma unroll
    for (int r = 0; r < 8; ++r) {
      p0[r] = __expf(p0[r] - m_new);
      p1[r] = __expf(p1[r] - m_new);
      lsum += p0[r] + p1[r];
    }
    lsum += __shfl_xor(lsum, 16, 32);
    l_run = l_run * rescale + lsum;

#pragma unroll
    for (int dt = 0; dt < 4; ++dt)
#pragma unroll
      for (int r = 0; r < 8; ++r) acc[dt][r] *= rescale;

    // Repack P^T into B-operand layout (lane=q; k = e + hf*16).
    v16h pB;
#pragma unroll
    for (int e = 0; e < 8; ++e) {
      const float o0 = __shfl_xor(p0[e], 16, 32);
      const float o1 = __shfl_xor(p1[e], 16, 32);
      pB[e]     = (_Float16)(hf ? o1 : p0[e]);
      pB[e + 8] = (_Float16)(hf ? p1[e] : o0);
    }

#pragma unroll
    for (int dt = 0; dt < 4; ++dt) {
      const _Float16* vrow = &Vw[(dt * 16 + l16) * 32 + hf * 16];
      acc[dt] = wmma_f16(pack_h8(*(const h8*)(vrow), *(const h8*)(vrow + 8)), pB, acc[dt]);
    }
    __syncthreads();   // release buf for the async writes issued in iteration kt+1
  }

  const float invl = (l_run > 0.0f) ? (1.0f / l_run) : 0.0f;
  _Float16* crow = ctx + ((size_t)b * LQ_ + q) * (H_ * DH_) + h * DH_;
#pragma unroll
  for (int dt = 0; dt < 4; ++dt) {
#pragma unroll
    for (int r = 0; r < 8; r += 2) {
      h2 v;
      v[0] = (_Float16)(acc[dt][r] * invl);
      v[1] = (_Float16)(acc[dt][r + 1] * invl);
      *(h2*)(crow + dt * 16 + hf * 8 + r) = v;
    }
  }
}

// ---------------------------------------------------------------------------
// Kernel 3: out = ctx(f16) @ Wo16^T -> f32.  Same 32x64/wave blocking.
// ---------------------------------------------------------------------------
__global__ void __launch_bounds__(256)
cpm3_out_proj(const _Float16* __restrict__ ctx, const _Float16* __restrict__ Wo16,
              float* __restrict__ out) {
  const int lane = threadIdx.x & 31;
  const int wave = threadIdx.x >> 5;
  const int l16 = lane & 15;
  const int hf  = lane >> 4;

  const int tile = blockIdx.x * 8 + wave;
  const int m0 = (tile >> 4) * 32;
  const int n0 = (tile & 15) * 64;

  const _Float16* __restrict__ Arow0 = ctx + (size_t)(m0 + l16) * D_;
  const _Float16* __restrict__ Arow1 = ctx + (size_t)(m0 + 16 + l16) * D_;
  const _Float16* __restrict__ Brow  = Wo16 + (size_t)(n0 + l16) * D_;

  v8f acc[2][4] = {};
#pragma unroll 2
  for (int kk = 0; kk < D_ / 32; ++kk) {
    const int ka = kk * 32 + hf * 16;
    const v16h A0 = pack_h8(*(const h8*)(Arow0 + ka), *(const h8*)(Arow0 + ka + 8));
    const v16h A1 = pack_h8(*(const h8*)(Arow1 + ka), *(const h8*)(Arow1 + ka + 8));
    const int kb = kk * 32 + hf * 16;
#pragma unroll
    for (int ni = 0; ni < 4; ++ni) {
      const v16h Bv = *(const v16h*)(Brow + (size_t)ni * 16 * D_ + kb);
      acc[0][ni] = wmma_f16(A0, Bv, acc[0][ni]);
      acc[1][ni] = wmma_f16(A1, Bv, acc[1][ni]);
    }
  }

#pragma unroll
  for (int mi = 0; mi < 2; ++mi)
#pragma unroll
    for (int ni = 0; ni < 4; ++ni) {
      const int n = n0 + ni * 16 + l16;
#pragma unroll
      for (int r = 0; r < 8; ++r) {
        const int m = m0 + mi * 16 + hf * 8 + r;
        out[(size_t)m * D_ + n] = acc[mi][ni][r];
      }
    }
}

// ---------------------------------------------------------------------------
extern "C" void kernel_launch(void* const* d_in, const int* in_sizes, int n_in,
                              void* d_out, int out_size, void* d_ws, size_t ws_size,
                              hipStream_t stream) {
  const float* query     = (const float*)d_in[0];
  const float* key_value = (const float*)d_in[1];
  const int*   mask      = (const int*)d_in[2];
  const float* bias      = (const float*)d_in[3];
  const float* Wq        = (const float*)d_in[4];
  const float* Wk        = (const float*)d_in[5];
  const float* Wv        = (const float*)d_in[6];
  const float* Wo        = (const float*)d_in[7];
  float* out = (float*)d_out;

  // ws layout (bytes): Qh|Kh|Vt|ctx (8MiB each) | Xq16|Xkv16 (8MiB each) | 4x W16 (2MiB each)
  const size_t MB = 1u << 20;
  char* ws = (char*)d_ws;
  _Float16* Qh    = (_Float16*)(ws);
  _Float16* Kh    = (_Float16*)(ws + 8 * MB);
  _Float16* Vt    = (_Float16*)(ws + 16 * MB);
  _Float16* ctx   = (_Float16*)(ws + 24 * MB);
  _Float16* Xq16  = (_Float16*)(ws + 32 * MB);
  _Float16* Xkv16 = (_Float16*)(ws + 40 * MB);
  _Float16* Wq16  = (_Float16*)(ws + 48 * MB);
  _Float16* Wk16  = (_Float16*)(ws + 50 * MB);
  _Float16* Wv16  = (_Float16*)(ws + 52 * MB);
  _Float16* Wo16  = (_Float16*)(ws + 54 * MB);

  (void)in_sizes; (void)n_in; (void)out_size; (void)ws_size;

  const int nX16 = (B_ * LQ_ * D_) / 16;   // 262144
  const int nW16 = (D_ * D_) / 16;         // 65536
  cpm3_cvt_f16<<<(nX16 + 255) / 256, 256, 0, stream>>>(query, Xq16, nX16);
  cpm3_cvt_f16<<<(nX16 + 255) / 256, 256, 0, stream>>>(key_value, Xkv16, nX16);
  cpm3_cvt_f16<<<(nW16 + 255) / 256, 256, 0, stream>>>(Wq, Wq16, nW16);
  cpm3_cvt_f16<<<(nW16 + 255) / 256, 256, 0, stream>>>(Wk, Wk16, nW16);
  cpm3_cvt_f16<<<(nW16 + 255) / 256, 256, 0, stream>>>(Wv, Wv16, nW16);
  cpm3_cvt_f16<<<(nW16 + 255) / 256, 256, 0, stream>>>(Wo, Wo16, nW16);

  cpm3_qkv_proj<<<dim3(256, 1, 3), 256, 0, stream>>>(Xq16, Xkv16, Wq16, Wk16, Wv16,
                                                     Qh, Kh, Vt);
  cpm3_attn<<<dim3(LQ_ / 128, B_ * H_), 256, 0, stream>>>(Qh, Kh, Vt, bias, mask, ctx);
  cpm3_out_proj<<<dim3(256, 1, 1), 256, 0, stream>>>(ctx, Wo16, out);
}
